// SelfAttention_61074434949807
// MI455X (gfx1250) — compile-verified, matrix operand order checked by reading, not run on hardware
//
#include <hip/hip_runtime.h>

typedef __attribute__((ext_vector_type(16))) _Float16 v16h;
typedef __attribute__((ext_vector_type(8)))  _Float16 v8h;
typedef __attribute__((ext_vector_type(8)))  float    v8f;
typedef int v4i __attribute__((vector_size(16)));

#define B_   2
#define D_   1024
#define N_   2048
#define H_   16
#define DA_  64
#define NEGV (-1e30f)

// Async global->LDS copy path (CDNA5 GLOBAL_LOAD_ASYNC_TO_LDS_B128), guarded so
// the file compiles on toolchains that lack the builtin.
#if defined(__has_builtin)
#if __has_builtin(__builtin_amdgcn_global_load_async_to_lds_b128) && \
    __has_builtin(__builtin_amdgcn_s_wait_asynccnt)
#define USE_ASYNC_LDS 1
#endif
#endif
#ifndef USE_ASYNC_LDS
#define USE_ASYNC_LDS 0
#endif

#if USE_ASYNC_LDS
typedef __attribute__((address_space(1))) v4i* gas_b128_t;
typedef __attribute__((address_space(3))) v4i* las_b128_t;
#endif

// Copy 16 bytes from global to LDS (per active lane).
static __device__ __forceinline__ void stage_b128(const _Float16* g, _Float16* l) {
#if USE_ASYNC_LDS
  __builtin_amdgcn_global_load_async_to_lds_b128((gas_b128_t)g, (las_b128_t)l, 0, 0);
#else
  *(v8h*)l = *(const v8h*)g;
#endif
}

static __device__ __forceinline__ void stage_wait() {
#if USE_ASYNC_LDS
  __builtin_amdgcn_s_wait_asynccnt(0);
#endif
}

// ---------------------------------------------------------------------------
// Fragment loader (wave32, 16-bit A/B 16x32 layout per CDNA5 ISA §7.12.2):
//   lane<16 : row = lane,    K = {0..7} and {16..23}
//   lane>=16: row = lane-16, K = {8..15} and {24..31}
// Requires K-contiguous storage -> two b128 loads per lane (global or LDS).
// ---------------------------------------------------------------------------
static __device__ __forceinline__ v16h load_frag(const _Float16* base, int stride) {
  int lane = threadIdx.x & 31;
  int r    = lane & 15;
  int koff = (lane >> 4) << 3;
  const _Float16* p = base + (size_t)r * stride + koff;
  v8h lo = *(const v8h*)(p);
  v8h hi = *(const v8h*)(p + 16);
  return __builtin_shufflevector(lo, hi, 0,1,2,3,4,5,6,7,8,9,10,11,12,13,14,15);
}

static __device__ __forceinline__ v8f wmma_f16(v16h a, v16h b, v8f c) {
  return __builtin_amdgcn_wmma_f32_16x16x32_f16(false, a, false, b, (short)0, c,
                                                false, false);
}

// Half-wave (16-lane group) reductions via ds_swizzle xor patterns.
static __device__ __forceinline__ float half_max(float v) {
  v = fmaxf(v, __builtin_bit_cast(float, __builtin_amdgcn_ds_swizzle(__builtin_bit_cast(int, v), 0x041f)));
  v = fmaxf(v, __builtin_bit_cast(float, __builtin_amdgcn_ds_swizzle(__builtin_bit_cast(int, v), 0x081f)));
  v = fmaxf(v, __builtin_bit_cast(float, __builtin_amdgcn_ds_swizzle(__builtin_bit_cast(int, v), 0x101f)));
  v = fmaxf(v, __builtin_bit_cast(float, __builtin_amdgcn_ds_swizzle(__builtin_bit_cast(int, v), 0x201f)));
  return v;
}
static __device__ __forceinline__ float half_sum(float v) {
  v += __builtin_bit_cast(float, __builtin_amdgcn_ds_swizzle(__builtin_bit_cast(int, v), 0x041f));
  v += __builtin_bit_cast(float, __builtin_amdgcn_ds_swizzle(__builtin_bit_cast(int, v), 0x081f));
  v += __builtin_bit_cast(float, __builtin_amdgcn_ds_swizzle(__builtin_bit_cast(int, v), 0x101f));
  v += __builtin_bit_cast(float, __builtin_amdgcn_ds_swizzle(__builtin_bit_cast(int, v), 0x201f));
  return v;
}

// ---------------------------------------------------------------------------
// Prep kernels: fp32 -> fp16 with layout changes for fragment-friendly loads.
// ---------------------------------------------------------------------------
__global__ void prep_x_kernel(const float* __restrict__ x, _Float16* __restrict__ xh) {
  int idx = blockIdx.x * blockDim.x + threadIdx.x;      // over B*N*D
  if (idx >= B_ * N_ * D_) return;
  int b = idx / (N_ * D_);
  int r = idx % (N_ * D_);
  int n = r / D_;
  int d = r % D_;
  xh[idx] = (_Float16)x[(size_t)b * D_ * N_ + (size_t)d * N_ + n];
}

__global__ void prep_wqkv_kernel(const float* __restrict__ Wq, const float* __restrict__ Wk,
                                 const float* __restrict__ Wv,
                                 _Float16* __restrict__ Wqt, _Float16* __restrict__ Wkt,
                                 _Float16* __restrict__ Wvt) {
  int idx = blockIdx.x * blockDim.x + threadIdx.x;      // over H*DA*D  (layout [h][e][d])
  if (idx >= H_ * DA_ * D_) return;
  int h = idx / (DA_ * D_);
  int r = idx % (DA_ * D_);
  int e = r / D_;
  int d = r % D_;
  size_t src = (size_t)h * D_ * DA_ + (size_t)d * DA_ + e;
  Wqt[idx] = (_Float16)Wq[src];
  Wkt[idx] = (_Float16)Wk[src];
  Wvt[idx] = (_Float16)Wv[src];
}

__global__ void prep_wo_kernel(const float* __restrict__ Wo, _Float16* __restrict__ Wot) {
  int idx = blockIdx.x * blockDim.x + threadIdx.x;      // over D*D  (Wot[n][d] = Wo[d][n])
  if (idx >= D_ * D_) return;
  int n = idx / D_;
  int d = idx % D_;
  Wot[idx] = (_Float16)Wo[(size_t)d * D_ + n];
}

// ---------------------------------------------------------------------------
// QKV projection: per (b,h), [128 x 1024] @ [1024 x 64] for q,k,v.
// 8 waves per WG, each wave owns 16 token rows. q scaled by 1/sqrt(Da).
// q,k stored [N][Da]; v stored transposed [Da][N] (B-fragment for P@V).
// ---------------------------------------------------------------------------
__global__ __launch_bounds__(256) void qkv_kernel(
    const _Float16* __restrict__ xh,
    const _Float16* __restrict__ Wqt, const _Float16* __restrict__ Wkt,
    const _Float16* __restrict__ Wvt,
    _Float16* __restrict__ qo, _Float16* __restrict__ ko, _Float16* __restrict__ vT) {
  const int NB = N_ / 128;
  int g  = blockIdx.x;
  int nb = g % NB;
  int bh = g / NB;
  int b  = bh / H_;
  int h  = bh % H_;
  int w    = threadIdx.x >> 5;
  int lane = threadIdx.x & 31;
  int n0   = nb * 128 + w * 16;

  const _Float16* xrow = xh  + (size_t)b * N_ * D_ + (size_t)n0 * D_;
  const _Float16* wq   = Wqt + (size_t)h * DA_ * D_;
  const _Float16* wk   = Wkt + (size_t)h * DA_ * D_;
  const _Float16* wv   = Wvt + (size_t)h * DA_ * D_;

  v8f aq[4] = {}, ak[4] = {}, av[4] = {};

  for (int kd = 0; kd < D_; kd += 32) {
    v16h A = load_frag(xrow + kd, D_);
#pragma unroll
    for (int t = 0; t < 4; ++t) {
      aq[t] = wmma_f16(A, load_frag(wq + (size_t)(t * 16) * D_ + kd, D_), aq[t]);
      ak[t] = wmma_f16(A, load_frag(wk + (size_t)(t * 16) * D_ + kd, D_), ak[t]);
      av[t] = wmma_f16(A, load_frag(wv + (size_t)(t * 16) * D_ + kd, D_), av[t]);
    }
  }

  size_t bhoff = ((size_t)b * H_ + h) * (size_t)N_ * DA_;
  int c  = lane & 15;
  int rb = (lane >> 4) << 3;
  const float qscale = 0.125f;   // 1/sqrt(64)
#pragma unroll
  for (int t = 0; t < 4; ++t) {
#pragma unroll
    for (int j = 0; j < 8; ++j) {
      int row = n0 + rb + j;
      qo[bhoff + (size_t)row * DA_ + t * 16 + c] = (_Float16)(aq[t][j] * qscale);
      ko[bhoff + (size_t)row * DA_ + t * 16 + c] = (_Float16)(ak[t][j]);
      vT[bhoff + (size_t)(t * 16 + c) * N_ + row] = (_Float16)(av[t][j]);
    }
  }
}

// ---------------------------------------------------------------------------
// Flash attention: 8 waves share one (b,h); per 32-key block the workgroup
// cooperatively stages the K tile (32x64) and V^T tile (64x32) into LDS
// (async global->LDS on CDNA5), then every wave computes:
//   4 score WMMAs -> masked online softmax (ds_swizzle half-wave reductions)
//   -> P transpose through LDS -> 4 P@V WMMAs into the O accumulators.
// ---------------------------------------------------------------------------
__global__ __launch_bounds__(256) void attn_kernel(
    const _Float16* __restrict__ qi, const _Float16* __restrict__ ki,
    const _Float16* __restrict__ vT, const float* __restrict__ mask,
    _Float16* __restrict__ heads) {
  __shared__ __align__(16) _Float16 kt[32 * DA_];       // [key][e]   4 KB
  __shared__ __align__(16) _Float16 vt[DA_ * 32];       // [e][key]   4 KB
  __shared__ __align__(16) _Float16 ptile[8][16 * 32];  // per-wave P 8 KB

  const int NB = N_ / 128;
  int g  = blockIdx.x;
  int nb = g % NB;
  int bh = g / NB;
  int b  = bh / H_;
  int h  = bh % H_;
  int tid  = threadIdx.x;
  int w    = tid >> 5;
  int lane = tid & 31;
  int n0   = nb * 128 + w * 16;

  size_t bhoff = ((size_t)b * H_ + h) * (size_t)N_ * DA_;
  const _Float16* qb = qi + bhoff;
  const _Float16* kb = ki + bhoff;
  const _Float16* vb = vT + bhoff;
  const float*  mrow = mask + (size_t)b * N_;

  // Q fragments for this wave's 16 rows (K = e in two 32-wide steps), kept live.
  v16h qa0 = load_frag(qb + (size_t)n0 * DA_ + 0,  DA_);
  v16h qa1 = load_frag(qb + (size_t)n0 * DA_ + 32, DA_);

  v8f O[4] = {};
  float m_run[8], l_run[8];
#pragma unroll
  for (int j = 0; j < 8; ++j) { m_run[j] = -3.0e38f; l_run[j] = 0.0f; }

  int c  = lane & 15;
  int rb = (lane >> 4) << 3;

  // Per-thread staging coordinates (one b128 each for K and V tiles).
  int kr = tid >> 3, kc = (tid & 7) << 3;   // K tile: 32 rows x 64 halves
  int vr = tid >> 2, vc = (tid & 3) << 3;   // V tile: 64 rows x 32 halves

  for (int k0 = 0; k0 < N_; k0 += 32) {
    // ---- cooperative stage of K/V tiles into LDS ----
    const _Float16* gk = kb + (size_t)(k0 + kr) * DA_ + kc;
    const _Float16* gv = vb + (size_t)vr * N_ + k0 + vc;
    stage_b128(gk, &kt[kr * DA_ + kc]);
    stage_b128(gv, &vt[vr * 32 + vc]);
    if (k0 + 32 < N_) {
      __builtin_prefetch(gk + 32 * DA_, 0, 1);   // next K tile
      __builtin_prefetch(gv + 32, 0, 1);         // next V tile
    }
    stage_wait();
    __syncthreads();

    // ---- scores: two 16x16 tiles, K = Da = 64 (two WMMA steps each) ----
    v8f S0 = {}, S1 = {};
    S0 = wmma_f16(qa0, load_frag(&kt[0], DA_), S0);
    S0 = wmma_f16(qa1, load_frag(&kt[32], DA_), S0);
    S1 = wmma_f16(qa0, load_frag(&kt[16 * DA_ + 0], DA_), S1);
    S1 = wmma_f16(qa1, load_frag(&kt[16 * DA_ + 32], DA_), S1);

    // ---- key mask (per lane: one key column per tile) ----
    float hm0 = mrow[k0 + c];
    float hm1 = mrow[k0 + 16 + c];
#pragma unroll
    for (int j = 0; j < 8; ++j) {
      S0[j] = S0[j] * hm0 + (1.0f - hm0) * NEGV;
      S1[j] = S1[j] * hm1 + (1.0f - hm1) * NEGV;
    }

    // ---- online softmax update, rows replicated across each 16-lane half ----
    float p0[8], p1[8];
#pragma unroll
    for (int j = 0; j < 8; ++j) {
      float mx    = half_max(fmaxf(S0[j], S1[j]));
      float mnew  = fmaxf(m_run[j], mx);
      float alpha = __expf(m_run[j] - mnew);
      float e0 = __expf(S0[j] - mnew);
      float e1 = __expf(S1[j] - mnew);
      float rs = half_sum(e0 + e1);
      l_run[j] = l_run[j] * alpha + rs;
      m_run[j] = mnew;
      p0[j] = e0;
      p1[j] = e1;
#pragma unroll
      for (int t = 0; t < 4; ++t) O[t][j] *= alpha;
    }

    // ---- transpose P (C layout -> A fragment) through per-wave LDS ----
#pragma unroll
    for (int j = 0; j < 8; ++j) {
      ptile[w][(rb + j) * 32 + c]      = (_Float16)p0[j];
      ptile[w][(rb + j) * 32 + 16 + c] = (_Float16)p1[j];
    }
    __syncthreads();
    v16h Pa = load_frag(&ptile[w][0], 32);

    // ---- O += P @ V  (B fragments from LDS vt[e][key], key-contiguous) ----
#pragma unroll
    for (int t = 0; t < 4; ++t)
      O[t] = wmma_f16(Pa, load_frag(&vt[(t * 16) * 32], 32), O[t]);
    __syncthreads();   // all waves done reading kt/vt/ptile before restage
  }

  // ---- finalize: O / l, apply query-side mask, store heads fp16 [N][D] ----
#pragma unroll
  for (int j = 0; j < 8; ++j) {
    int row = n0 + rb + j;
    float vm  = mrow[row];
    float inv = (l_run[j] > 0.0f) ? (vm / l_run[j]) : 0.0f;
#pragma unroll
    for (int t = 0; t < 4; ++t) {
      heads[((size_t)b * N_ + row) * D_ + h * DA_ + t * 16 + c] =
          (_Float16)(O[t][j] * inv);
    }
  }
}

// ---------------------------------------------------------------------------
// Output projection: [N x 1024] @ [1024 x 1024], fused transpose-store to
// (B, D, N) in fp32. Each wave: 16 rows x 128 cols (8 accumulator tiles).
// ---------------------------------------------------------------------------
__global__ __launch_bounds__(256) void out_kernel(
    const _Float16* __restrict__ heads, const _Float16* __restrict__ Wot,
    float* __restrict__ out) {
  const int NB = N_ / 128, DB = D_ / 128;
  int g  = blockIdx.x;
  int db = g % DB;
  int r  = g / DB;
  int nb = r % NB;
  int b  = r / NB;
  int w    = threadIdx.x >> 5;
  int lane = threadIdx.x & 31;
  int n0   = nb * 128 + w * 16;
  int c0   = db * 128;

  const _Float16* hb = heads + ((size_t)b * N_ + n0) * D_;
  const _Float16* wb = Wot + (size_t)c0 * D_;

  v8f acc[8] = {};
  for (int kd = 0; kd < D_; kd += 32) {
    v16h A = load_frag(hb + kd, D_);
#pragma unroll
    for (int t = 0; t < 8; ++t)
      acc[t] = wmma_f16(A, load_frag(wb + (size_t)(t * 16) * D_ + kd, D_), acc[t]);
  }

  int cc = lane & 15;
  int rb = (lane >> 4) << 3;
#pragma unroll
  for (int t = 0; t < 8; ++t) {
#pragma unroll
    for (int j = 0; j < 8; ++j) {
      int d = c0 + t * 16 + cc;
      int n = n0 + rb + j;
      out[(size_t)b * D_ * N_ + (size_t)d * N_ + n] = acc[t][j];
    }
  }
}

// ---------------------------------------------------------------------------
extern "C" void kernel_launch(void* const* d_in, const int* in_sizes, int n_in,
                              void* d_out, int out_size, void* d_ws, size_t ws_size,
                              hipStream_t stream) {
  const float* x    = (const float*)d_in[0];
  const float* mask = (const float*)d_in[1];
  const float* Wq   = (const float*)d_in[2];
  const float* Wk   = (const float*)d_in[3];
  const float* Wv   = (const float*)d_in[4];
  const float* Wo   = (const float*)d_in[5];
  float* out = (float*)d_out;

  char* ws = (char*)d_ws;
  _Float16* xh   = (_Float16*)ws; ws += (size_t)B_ * N_ * D_ * 2;
  _Float16* Wqt  = (_Float16*)ws; ws += (size_t)H_ * DA_ * D_ * 2;
  _Float16* Wkt  = (_Float16*)ws; ws += (size_t)H_ * DA_ * D_ * 2;
  _Float16* Wvt  = (_Float16*)ws; ws += (size_t)H_ * DA_ * D_ * 2;
  _Float16* Wot  = (_Float16*)ws; ws += (size_t)D_ * D_ * 2;
  _Float16* qb   = (_Float16*)ws; ws += (size_t)B_ * H_ * N_ * DA_ * 2;
  _Float16* kb   = (_Float16*)ws; ws += (size_t)B_ * H_ * N_ * DA_ * 2;
  _Float16* vT   = (_Float16*)ws; ws += (size_t)B_ * H_ * N_ * DA_ * 2;
  _Float16* hbuf = (_Float16*)ws; ws += (size_t)B_ * N_ * D_ * 2;

  prep_x_kernel<<<(B_ * N_ * D_ + 255) / 256, 256, 0, stream>>>(x, xh);
  prep_wqkv_kernel<<<(H_ * DA_ * D_ + 255) / 256, 256, 0, stream>>>(Wq, Wk, Wv,
                                                                    Wqt, Wkt, Wvt);
  prep_wo_kernel<<<(D_ * D_ + 255) / 256, 256, 0, stream>>>(Wo, Wot);

  qkv_kernel<<<B_ * H_ * (N_ / 128), 256, 0, stream>>>(xh, Wqt, Wkt, Wvt, qb, kb, vT);
  attn_kernel<<<B_ * H_ * (N_ / 128), 256, 0, stream>>>(qb, kb, vT, mask, hbuf);
  out_kernel<<<B_ * (N_ / 128) * (D_ / 128), 256, 0, stream>>>(hbuf, Wot, out);
}
